// EGNNModel_13700945674926
// MI455X (gfx1250) — matrix operand order repaired and checked by reading, not compiled, not run on hardware
//
#include <hip/hip_runtime.h>
#include <hip/hip_bf16.h>
#include <stdint.h>

#define DD   128
#define TE   64     // edge/node rows per block
#define NTH  128    // 4 waves of 32
#define LN_EPS 1e-5f

typedef __attribute__((ext_vector_type(16))) _Float16 v16h;
typedef __attribute__((ext_vector_type(8)))  float    v8f;

// ---------------------------------------------------------------------------
// CDNA5 async global->LDS copy (ASYNCcnt-tracked). The low 32 bits of a
// generic shared-memory pointer are the LDS byte offset (ISA aperture rule).
// ---------------------------------------------------------------------------
__device__ __forceinline__ void async_copy_b128(void* lds_dst, const void* gsrc) {
  uint32_t ldsOff = (uint32_t)(uintptr_t)lds_dst;
  asm volatile("global_load_async_to_lds_b128 %0, %1, off"
               :
               : "v"(ldsOff), "v"(gsrc)
               : "memory");
}
__device__ __forceinline__ void wait_async0() {
  asm volatile("s_wait_asynccnt 0" ::: "memory");
}

// ---------------------------------------------------------------------------
// WMMA fragment helpers (CDNA5 16x16x32 f16, wave32)
// A layout (16-bit A 16x32): lane m=lane&15, half=lane>>4;
//   vgpr v -> K pair at 2v + 16*half (element e -> K = 16*half + e, matches B)
// B layout (32x16): lane n=lane&15, half=lane>>4; element e -> K = 16*half + e
// C layout: lane n=lane&15; vgpr r -> row M = r + 8*half
// ---------------------------------------------------------------------------
__device__ __forceinline__ v16h ldsA_frag(const uint32_t* row, int kc, int half) {
  union { uint32_t u[8]; v16h h; } a;
  const int base = kc * 16 + 8 * half;
#pragma unroll
  for (int v = 0; v < 4; ++v) a.u[v] = row[base + v];
#pragma unroll
  for (int v = 0; v < 4; ++v) a.u[4 + v] = row[base + 4 + v];
  return a.h;
}

__device__ __forceinline__ v16h gB_frag(const _Float16* __restrict__ wp, int kc, int nt, int lane) {
  union { uint4 q[2]; v16h h; } b;
  const uint4* p = (const uint4*)(wp + ((size_t)((kc * 8 + nt) * 32 + lane)) * 16);
  b.q[0] = p[0];
  b.q[1] = p[1];
  return b.h;
}

// Double-buffered GEMM: one B (and A at tile boundary) load always in flight
// ahead of the WMMA so the matrix pipe is not serialized on L2 latency.
template <int KT>
__device__ __forceinline__ void wave_gemm(const uint32_t* aRow, const _Float16* __restrict__ wp,
                                          int lane, v8f acc[8]) {
  const int half = lane >> 4;
  v16h a = ldsA_frag(aRow, 0, half);
  v16h b = gB_frag(wp, 0, 0, lane);
#pragma unroll
  for (int kc = 0; kc < KT; ++kc) {
#pragma unroll
    for (int nt = 0; nt < 8; ++nt) {
      v16h bn = b, an = a;
      if (nt < 7) {
        bn = gB_frag(wp, kc, nt + 1, lane);
      } else if (kc + 1 < KT) {
        bn = gB_frag(wp, kc + 1, 0, lane);
        an = ldsA_frag(aRow, kc + 1, half);
      }
      acc[nt] = __builtin_amdgcn_wmma_f32_16x16x32_f16(
          /*neg_a=*/false, a, /*neg_b=*/false, b,
          /*c_mod=*/(short)0, acc[nt], /*reuse_a=*/false, /*reuse_b=*/false);
      a = an;
      b = bn;
    }
  }
}

__device__ __forceinline__ void store_acc(float (*sM)[DD], int eb, int lane, const v8f acc[8]) {
  const int n0 = lane & 15, half = lane >> 4;
#pragma unroll
  for (int nt = 0; nt < 8; ++nt)
#pragma unroll
    for (int r = 0; r < 8; ++r)
      sM[eb + r + 8 * half][nt * 16 + n0] = acc[nt][r];
}

// Wave-level LayerNorm (+bias folded in via `pre`) + ReLU -> f16 row.
// 2 lanes per row, 64 dims each, one shfl_xor(16) for the cross-half reduce.
template <typename PreF>
__device__ __forceinline__ void ln_relu_to_f16(PreF pre, const float* __restrict__ g,
                                               const float* __restrict__ be, int half,
                                               _Float16* __restrict__ outRow) {
  float s1 = 0.f, s2 = 0.f;
#pragma unroll 8
  for (int k = 0; k < 64; ++k) {
    float v = pre(half * 64 + k);
    s1 += v;
    s2 += v * v;
  }
  s1 += __shfl_xor(s1, 16);
  s2 += __shfl_xor(s2, 16);
  float mean = s1 * (1.f / 128.f);
  float inv = rsqrtf(s2 * (1.f / 128.f) - mean * mean + LN_EPS);
#pragma unroll 8
  for (int k = 0; k < 64; ++k) {
    int kk = half * 64 + k;
    outRow[kk] = (_Float16)fmaxf((pre(kk) - mean) * inv * g[kk] + be[kk], 0.f);
  }
}

// ---------------------------------------------------------------------------
// Weight packer: W[K][128] fp32 row-major -> f16 B-fragment stream
// ---------------------------------------------------------------------------
__global__ void pack_b_kernel(const float* __restrict__ W, _Float16* __restrict__ out, int K) {
  int idx = blockIdx.x * blockDim.x + threadIdx.x;
  int total = (K / 32) * 8 * 32;
  if (idx >= total) return;
  int lane = idx & 31;
  int nt = (idx >> 5) & 7;
  int kc = idx >> 8;
  int n = nt * 16 + (lane & 15);
  int kbase = kc * 32 + (lane >> 4) * 16;
  _Float16* dst = out + (size_t)idx * 16;
#pragma unroll
  for (int e = 0; e < 16; ++e) dst[e] = (_Float16)W[(size_t)(kbase + e) * DD + n];
}

// ---------------------------------------------------------------------------
// h = emb[atoms] @ proj_w + proj_b ; writes f32 h and f16 mirror h16
// ---------------------------------------------------------------------------
__global__ __launch_bounds__(NTH) void egnn_embed_kernel(
    const int* __restrict__ atoms, const float* __restrict__ emb,
    const _Float16* __restrict__ projp, const float* __restrict__ projb,
    float* __restrict__ h, _Float16* __restrict__ h16, int N) {
  __shared__ __align__(16) _Float16 sA[TE][DD];
  const int tid = threadIdx.x, lane = tid & 31, n0 = lane & 15, half = lane >> 4;
  const int eb = (tid >> 5) * 16;
  const long nbase = (long)blockIdx.x * TE;

  for (int idx = tid; idx < TE * DD; idx += NTH) {
    int e = idx >> 7, k = idx & 127;
    long node = nbase + e;
    float v = 0.f;
    if (node < N) v = emb[(long)atoms[node] * DD + k];
    sA[e][k] = (_Float16)v;
  }
  __syncthreads();

  v8f acc[8];
#pragma unroll
  for (int nt = 0; nt < 8; ++nt) acc[nt] = (v8f){};
  wave_gemm<4>((const uint32_t*)&sA[eb + n0][0], projp, lane, acc);

#pragma unroll
  for (int nt = 0; nt < 8; ++nt) {
    float bv = projb[nt * 16 + n0];
#pragma unroll
    for (int r = 0; r < 8; ++r) {
      long node = nbase + eb + r + 8 * half;
      if (node < N) {
        float v = acc[nt][r] + bv;
        h[node * DD + nt * 16 + n0] = v;
        h16[node * DD + nt * 16 + n0] = (_Float16)v;
      }
    }
  }
}

// ---------------------------------------------------------------------------
// Edge kernel: async-gather f16 h rows into LDS, msg1->LN->msg2->LN, scatter m,
// pos MLP -> per-edge scalar, scatter diff*w + counts.
// ---------------------------------------------------------------------------
__global__ __launch_bounds__(NTH) void egnn_edge_kernel(
    const _Float16* __restrict__ h16, const float* __restrict__ pos,
    const int* __restrict__ srcI, const int* __restrict__ dstI, int E,
    const _Float16* __restrict__ w1p, const float* __restrict__ w1last,
    const float* __restrict__ b1, const float* __restrict__ g1, const float* __restrict__ be1,
    const _Float16* __restrict__ w2p,
    const float* __restrict__ b2, const float* __restrict__ g2, const float* __restrict__ be2,
    const _Float16* __restrict__ pw1p,
    const float* __restrict__ pb1, const float* __restrict__ pg, const float* __restrict__ pbe,
    const float* __restrict__ pw2, const float* __restrict__ pb2,
    float* __restrict__ msg_aggr, float* __restrict__ cnt, float* __restrict__ pos_aggr) {
  // 32KB A (aliased by the two 16KB f16 staging buffers) + 32KB f32 GEMM-out
  __shared__ __align__(16) uint8_t smem[TE * 2 * DD * 2 + TE * DD * 4];
  _Float16(*sA)[2 * DD] = (_Float16(*)[2 * DD])smem;
  _Float16(*sX)[DD] = (_Float16(*)[DD])smem;                   // aliases sA rows 0..31
  _Float16(*sY)[DD] = (_Float16(*)[DD])(smem + TE * DD * 2);   // aliases sA rows 32..63
  float(*sM)[DD] = (float(*)[DD])(smem + TE * 2 * DD * 2);
  __shared__ int sS[TE], sN[TE];
  __shared__ float sDist[TE], sDx[TE], sDy[TE], sDz[TE];

  const int tid = threadIdx.x, lane = tid & 31, n0 = lane & 15, half = lane >> 4;
  const int eb = (tid >> 5) * 16;
  const long ebase = (long)blockIdx.x * TE;

  if (tid < TE) {
    long e = ebase + tid;
    int s = 0, d = 0;
    float dx = 0.f, dy = 0.f, dz = 0.f, di = 0.f;
    if (e < E) {
      s = srcI[e];
      d = dstI[e];
      dx = pos[(long)d * 3 + 0] - pos[(long)s * 3 + 0];
      dy = pos[(long)d * 3 + 1] - pos[(long)s * 3 + 1];
      dz = pos[(long)d * 3 + 2] - pos[(long)s * 3 + 2];
      di = sqrtf(fmaxf(dx * dx + dy * dy + dz * dz, 1e-12f));
    }
    sS[tid] = s; sN[tid] = d;
    sDx[tid] = dx; sDy[tid] = dy; sDz[tid] = dz; sDist[tid] = di;
  }
  __syncthreads();

  // async gather concat(h16[dst], h16[src]) -> LDS; 16B per op, 16 ops/thread
  for (int idx = tid; idx < TE * 32; idx += NTH) {
    int e = idx >> 5;
    int seg = idx & 31;                    // 0..15: dst row, 16..31: src row
    int node = (seg < 16) ? sN[e] : sS[e]; // invalid edges clamped to node 0
    async_copy_b128(&sA[e][seg * 8], h16 + (size_t)node * DD + (seg & 15) * 8);
  }
  wait_async0();
  __syncthreads();

  v8f acc[8];

  // ---- msg layer 1: [16 x 257] @ [257 x 128] (bias + dist column folded into LN) ----
#pragma unroll
  for (int nt = 0; nt < 8; ++nt) acc[nt] = (v8f){};
  wave_gemm<8>((const uint32_t*)&sA[eb + n0][0], w1p, lane, acc);
  store_acc(sM, eb, lane, acc);
  __syncthreads();

  {
    const float* row = &sM[eb + n0][0];
    const float di = sDist[eb + n0];
    auto pre = [&](int kk) { return row[kk] + b1[kk] + di * w1last[kk]; };
    ln_relu_to_f16(pre, g1, be1, half, &sX[eb + n0][0]);
  }
  __syncthreads();

  // ---- msg layer 2: [16 x 128] @ [128 x 128] ----
#pragma unroll
  for (int nt = 0; nt < 8; ++nt) acc[nt] = (v8f){};
  wave_gemm<4>((const uint32_t*)&sX[eb + n0][0], w2p, lane, acc);
  store_acc(sM, eb, lane, acc);
  __syncthreads();

  {
    const float* row = &sM[eb + n0][0];
    auto pre = [&](int kk) { return row[kk] + b2[kk]; };
    ln_relu_to_f16(pre, g2, be2, half, &sY[eb + n0][0]);
  }
  __syncthreads();

  // scatter m into msg_aggr (segment-sum over dst)
  for (int idx = tid; idx < TE * DD; idx += NTH) {
    int e = idx >> 7, k = idx & 127;
    if (ebase + e < E) atomicAdd(&msg_aggr[(long)sN[e] * DD + k], (float)sY[e][k]);
  }

  // ---- pos MLP: w = relu(LN(m @ pos_w1 + pb1)) . pos_w2 + pb2 ----
#pragma unroll
  for (int nt = 0; nt < 8; ++nt) acc[nt] = (v8f){};
  wave_gemm<4>((const uint32_t*)&sY[eb + n0][0], pw1p, lane, acc);
  store_acc(sM, eb, lane, acc);
  __syncthreads();

  {
    const float* row = &sM[eb + n0][0];
    auto pre = [&](int kk) { return row[kk] + pb1[kk]; };
    float s1 = 0.f, s2 = 0.f;
#pragma unroll 8
    for (int k = 0; k < 64; ++k) {
      float v = pre(half * 64 + k);
      s1 += v;
      s2 += v * v;
    }
    s1 += __shfl_xor(s1, 16);
    s2 += __shfl_xor(s2, 16);
    float mean = s1 * (1.f / 128.f);
    float inv = rsqrtf(s2 * (1.f / 128.f) - mean * mean + LN_EPS);
    float dot = 0.f;
#pragma unroll 8
    for (int k = 0; k < 64; ++k) {
      int kk = half * 64 + k;
      float v = fmaxf((pre(kk) - mean) * inv * pg[kk] + pbe[kk], 0.f);
      dot += v * pw2[kk];
    }
    dot += __shfl_xor(dot, 16);
    float wgt = dot + pb2[0];
    int e = eb + n0;
    if (half == 0 && ebase + e < E) {
      int node = sN[e];
      atomicAdd(&pos_aggr[(long)node * 3 + 0], sDx[e] * wgt);
      atomicAdd(&pos_aggr[(long)node * 3 + 1], sDy[e] * wgt);
      atomicAdd(&pos_aggr[(long)node * 3 + 2], sDz[e] * wgt);
      atomicAdd(&cnt[node], 1.f);
    }
  }
}

// ---------------------------------------------------------------------------
// Node kernel: u = relu(LN(relu(LN(cat(h,msg)@W1))@W2)); h += u (f32 + f16
// mirror); pos += pos_aggr / max(cnt,1)
// ---------------------------------------------------------------------------
__global__ __launch_bounds__(NTH) void egnn_node_kernel(
    float* __restrict__ h, _Float16* __restrict__ h16, float* __restrict__ pos,
    const float* __restrict__ msg_aggr, const float* __restrict__ cnt,
    const float* __restrict__ pos_aggr, int N,
    const _Float16* __restrict__ uw1p,
    const float* __restrict__ ub1, const float* __restrict__ ug1, const float* __restrict__ ube1,
    const _Float16* __restrict__ uw2p,
    const float* __restrict__ ub2, const float* __restrict__ ug2, const float* __restrict__ ube2) {
  __shared__ __align__(16) uint8_t smem[TE * 2 * DD * 2 + TE * DD * 4];
  _Float16(*sA)[2 * DD] = (_Float16(*)[2 * DD])smem;
  _Float16(*sX)[DD] = (_Float16(*)[DD])smem;  // aliases sA low half
  float(*sM)[DD] = (float(*)[DD])(smem + TE * 2 * DD * 2);

  const int tid = threadIdx.x, lane = tid & 31, n0 = lane & 15, half = lane >> 4;
  const int eb = (tid >> 5) * 16;
  const long nbase = (long)blockIdx.x * TE;

  // h half: async copy of f16 rows; msg half: f32 -> f16 convert
  for (int idx = tid; idx < TE * 16; idx += NTH) {
    int e = idx >> 4, seg = idx & 15;
    long node = nbase + e;
    long cl = (node < N) ? node : 0;
    async_copy_b128(&sA[e][seg * 8], h16 + (size_t)cl * DD + seg * 8);
  }
  for (int idx = tid; idx < TE * DD; idx += NTH) {
    int e = idx >> 7, k = idx & 127;
    long node = nbase + e;
    float v = 0.f;
    if (node < N) v = msg_aggr[node * DD + k];
    sA[e][DD + k] = (_Float16)v;
  }
  wait_async0();
  __syncthreads();

  v8f acc[8];
#pragma unroll
  for (int nt = 0; nt < 8; ++nt) acc[nt] = (v8f){};
  wave_gemm<8>((const uint32_t*)&sA[eb + n0][0], uw1p, lane, acc);
  store_acc(sM, eb, lane, acc);
  __syncthreads();

  {
    const float* row = &sM[eb + n0][0];
    auto pre = [&](int kk) { return row[kk] + ub1[kk]; };
    ln_relu_to_f16(pre, ug1, ube1, half, &sX[eb + n0][0]);
  }
  __syncthreads();

#pragma unroll
  for (int nt = 0; nt < 8; ++nt) acc[nt] = (v8f){};
  wave_gemm<4>((const uint32_t*)&sX[eb + n0][0], uw2p, lane, acc);
  store_acc(sM, eb, lane, acc);
  __syncthreads();

  {  // LN + ReLU + residual to f32 h and f16 mirror
    long node = nbase + eb + n0;
    const float* row = &sM[eb + n0][0];
    auto pre = [&](int kk) { return row[kk] + ub2[kk]; };
    float s1 = 0.f, s2 = 0.f;
#pragma unroll 8
    for (int k = 0; k < 64; ++k) {
      float v = pre(half * 64 + k);
      s1 += v;
      s2 += v * v;
    }
    s1 += __shfl_xor(s1, 16);
    s2 += __shfl_xor(s2, 16);
    float mean = s1 * (1.f / 128.f);
    float inv = rsqrtf(s2 * (1.f / 128.f) - mean * mean + LN_EPS);
    if (node < N) {
#pragma unroll 8
      for (int k = 0; k < 64; ++k) {
        int kk = half * 64 + k;
        float u = fmaxf((pre(kk) - mean) * inv * ug2[kk] + ube2[kk], 0.f);
        float nv = h[node * DD + kk] + u;
        h[node * DD + kk] = nv;
        h16[node * DD + kk] = (_Float16)nv;
      }
    }
  }

  if (tid < TE) {
    long node = nbase + tid;
    if (node < N) {
      float c = fmaxf(cnt[node], 1.f);
      pos[node * 3 + 0] += pos_aggr[node * 3 + 0] / c;
      pos[node * 3 + 1] += pos_aggr[node * 3 + 1] / c;
      pos[node * 3 + 2] += pos_aggr[node * 3 + 2] / c;
    }
  }
}

// ---------------------------------------------------------------------------
// global_add_pool + pred MLP
// ---------------------------------------------------------------------------
__global__ void pool_kernel(const float* __restrict__ h, const int* __restrict__ batch,
                            float* __restrict__ pooled, int N) {
  long idx = (long)blockIdx.x * blockDim.x + threadIdx.x;
  if (idx >= (long)N * DD) return;
  int node = (int)(idx >> 7);
  int k = (int)(idx & 127);
  atomicAdd(&pooled[(long)batch[node] * DD + k], h[idx]);
}

__global__ __launch_bounds__(64) void pred_kernel(
    const float* __restrict__ pooled, const float* __restrict__ w1, const float* __restrict__ b1,
    const float* __restrict__ w2, const float* __restrict__ b2, float* __restrict__ out, int G) {
  __shared__ float sp[64][DD];
  int tid = threadIdx.x;
  for (int idx = tid; idx < G * DD; idx += 64) sp[idx >> 7][idx & 127] = pooled[idx];
  __syncthreads();
  if (tid >= G) return;
  float acc = b2[0];
  for (int j = 0; j < DD; ++j) {
    float s = b1[j];
    for (int k = 0; k < DD; ++k) s += sp[tid][k] * w1[(long)k * DD + j];
    acc += fmaxf(s, 0.f) * w2[j];
  }
  out[tid] = acc;
}

// ---------------------------------------------------------------------------
// Host launcher
// ---------------------------------------------------------------------------
extern "C" void kernel_launch(void* const* d_in, const int* in_sizes, int n_in,
                              void* d_out, int out_size, void* d_ws, size_t ws_size,
                              hipStream_t stream) {
  const int* atoms = (const int*)d_in[0];
  const float* pos_in = (const float*)d_in[1];
  const int* edge_index = (const int*)d_in[2];
  const int* batch = (const int*)d_in[3];
  const float* emb = (const float*)d_in[4];
  const float* proj_w = (const float*)d_in[5];
  const float* proj_b = (const float*)d_in[6];
  const int N = in_sizes[0];
  const int E = in_sizes[2] / 2;
  const int G = out_size;
  const int L = 5;
  const int* srcI = edge_index;        // row 0 = source j
  const int* dstI = edge_index + E;    // row 1 = target i

  auto LP = [&](int l, int j) { return (const float*)d_in[7 + l * 22 + j]; };
  const float* pred_w1 = (const float*)d_in[7 + L * 22 + 0];
  const float* pred_b1 = (const float*)d_in[7 + L * 22 + 1];
  const float* pred_w2 = (const float*)d_in[7 + L * 22 + 2];
  const float* pred_b2 = (const float*)d_in[7 + L * 22 + 3];

  // workspace carving (256B aligned)
  uint8_t* wsp = (uint8_t*)d_ws;
  auto carve = [&](size_t bytes) {
    uint8_t* p = wsp;
    wsp += (bytes + 255) & ~(size_t)255;
    return p;
  };
  float* h = (float*)carve((size_t)N * DD * 4);
  _Float16* h16 = (_Float16*)carve((size_t)N * DD * 2);
  float* msg_aggr = (float*)carve((size_t)N * DD * 4);
  float* cnt = (float*)carve((size_t)N * 4);
  float* pos_aggr = (float*)carve((size_t)N * 3 * 4);
  float* posbuf = (float*)carve((size_t)N * 3 * 4);
  float* pooled = (float*)carve((size_t)G * DD * 4);
  _Float16* projp = (_Float16*)carve((size_t)128 * 128 * 2);
  _Float16* lwp[5][5];
  for (int l = 0; l < L; ++l) {
    lwp[l][0] = (_Float16*)carve((size_t)256 * 128 * 2);  // msg_w1 (first 256 rows)
    lwp[l][1] = (_Float16*)carve((size_t)128 * 128 * 2);  // msg_w2
    lwp[l][2] = (_Float16*)carve((size_t)128 * 128 * 2);  // pos_w1
    lwp[l][3] = (_Float16*)carve((size_t)256 * 128 * 2);  // upd_w1
    lwp[l][4] = (_Float16*)carve((size_t)128 * 128 * 2);  // upd_w2
  }

  // pack weights into WMMA B-fragment layout (cheap; rerun every call)
  pack_b_kernel<<<128 / 32, 256, 0, stream>>>(proj_w, projp, 128);
  for (int l = 0; l < L; ++l) {
    pack_b_kernel<<<256 / 32, 256, 0, stream>>>(LP(l, 0), lwp[l][0], 256);
    pack_b_kernel<<<128 / 32, 256, 0, stream>>>(LP(l, 4), lwp[l][1], 128);
    pack_b_kernel<<<128 / 32, 256, 0, stream>>>(LP(l, 8), lwp[l][2], 128);
    pack_b_kernel<<<256 / 32, 256, 0, stream>>>(LP(l, 14), lwp[l][3], 256);
    pack_b_kernel<<<128 / 32, 256, 0, stream>>>(LP(l, 18), lwp[l][4], 128);
  }

  hipMemcpyAsync(posbuf, pos_in, (size_t)N * 3 * 4, hipMemcpyDeviceToDevice, stream);

  const int nodeBlocks = (N + TE - 1) / TE;
  const int edgeBlocks = (E + TE - 1) / TE;

  egnn_embed_kernel<<<nodeBlocks, NTH, 0, stream>>>(atoms, emb, projp, proj_b, h, h16, N);

  for (int l = 0; l < L; ++l) {
    hipMemsetAsync(msg_aggr, 0, (size_t)N * DD * 4, stream);
    hipMemsetAsync(cnt, 0, (size_t)N * 4, stream);
    hipMemsetAsync(pos_aggr, 0, (size_t)N * 3 * 4, stream);

    const float* w1last = LP(l, 0) + (size_t)256 * 128;  // dist row of msg_w1
    egnn_edge_kernel<<<edgeBlocks, NTH, 0, stream>>>(
        h16, posbuf, srcI, dstI, E,
        lwp[l][0], w1last, LP(l, 1), LP(l, 2), LP(l, 3),
        lwp[l][1], LP(l, 5), LP(l, 6), LP(l, 7),
        lwp[l][2], LP(l, 9), LP(l, 10), LP(l, 11),
        LP(l, 12), LP(l, 13),
        msg_aggr, cnt, pos_aggr);

    egnn_node_kernel<<<nodeBlocks, NTH, 0, stream>>>(
        h, h16, posbuf, msg_aggr, cnt, pos_aggr, N,
        lwp[l][3], LP(l, 15), LP(l, 16), LP(l, 17),
        lwp[l][4], LP(l, 19), LP(l, 20), LP(l, 21));
  }

  hipMemsetAsync(pooled, 0, (size_t)G * DD * 4, stream);
  {
    long total = (long)N * DD;
    int blocks = (int)((total + 255) / 256);
    pool_kernel<<<blocks, 256, 0, stream>>>(h, batch, pooled, N);
  }
  pred_kernel<<<1, 64, 0, stream>>>(pooled, pred_w1, pred_b1, pred_w2, pred_b2,
                                    (float*)d_out, G);
}
